// UnsupervisedGAT_9174050144733
// MI455X (gfx1250) — compile-verified, matrix operand order checked by reading, not run on hardware
//
#include <hip/hip_runtime.h>
#include <hip/hip_bf16.h>

// Problem constants (match reference)
#define NN 50000
#define EE 800000
#define HH 8
#define DH 16
#define DF 128          // D_IN == D_HID == 128
#define LRELU_ATTN 0.2f
#define LRELU_ACT 0.01f

typedef __attribute__((ext_vector_type(16))) __bf16          v16bf;
typedef __attribute__((ext_vector_type(8)))  float           v8f;
typedef __attribute__((ext_vector_type(8)))  unsigned short  u16x8;
typedef __attribute__((ext_vector_type(16))) unsigned short  u16x16;

// f32 -> bf16 bits, round-to-nearest-even
__device__ __forceinline__ unsigned short f2bf_bits(float f) {
    unsigned u = __float_as_uint(f);
    unsigned r = u + 0x7FFFu + ((u >> 16) & 1u);
    return (unsigned short)(r >> 16);
}

// order-preserving float atomic max (init memory to -inf bits 0xFF800000)
__device__ __forceinline__ void atomicMaxF(float* addr, float v) {
    if (v >= 0.0f) atomicMax((int*)addr, __float_as_int(v));
    else           atomicMin((unsigned int*)addr, __float_as_uint(v));
}

// butterfly add over 16-lane groups via ds_swizzle (group-of-32 XOR mode):
// imm = (xor_mask << 10) | 0x1f   (and_mask=0x1f, or_mask=0)
template <int IMM>
__device__ __forceinline__ float swz_add(float v) {
    return v + __int_as_float(__builtin_amdgcn_ds_swizzle(__float_as_int(v), IMM));
}
__device__ __forceinline__ float red16(float v) {
    v = swz_add<0x041f>(v);   // xor 1
    v = swz_add<0x081f>(v);   // xor 2
    v = swz_add<0x101f>(v);   // xor 4
    v = swz_add<0x201f>(v);   // xor 8
    return v;
}

// ---------------------------------------------------------------------------
// Fill a u32 pattern (used for -inf init, zero init)
__global__ __launch_bounds__(256) void fill_u32(unsigned* p, size_t n, unsigned pat) {
    size_t i = (size_t)blockIdx.x * 256u + threadIdx.x;
    if (i < n) p[i] = pat;
}

// ---------------------------------------------------------------------------
// Transpose + convert W [K=128][N=128] f32 -> Wt [N][K] bf16 (K contiguous)
__global__ __launch_bounds__(256) void conv_wt(const float* __restrict__ W,
                                               unsigned short* __restrict__ Wt) {
    int i = blockIdx.x * 256 + threadIdx.x;
    if (i < DF * DF) {
        int n = i >> 7, k = i & 127;
        Wt[n * DF + k] = f2bf_bits(W[k * DF + n]);
    }
}

// ---------------------------------------------------------------------------
// Fused GEMM (h = X @ W, bf16 WMMA, f32 accum) + attention projections.
// Block = 256 threads = 8 waves; block computes 16 rows; wave w computes the
// 16x16 tile for columns [16w,16w+16) == head w, then reduces el/er in-wave.
// Weights are async-DMA'd into LDS (global_load_async_to_lds_b128) while the
// A tile is converted f32->bf16 into LDS; hot loop is ds_load + 4x WMMA.
__global__ __launch_bounds__(256) void gat_gemm_wmma(
    const float* __restrict__ X, const unsigned short* __restrict__ Wt,
    const float* __restrict__ al, const float* __restrict__ ar,
    float* __restrict__ hout, float* __restrict__ el, float* __restrict__ er)
{
    // padded rows: 136 elems * 2B = 272B stride (spreads LDS banks)
    __shared__ __align__(16) unsigned short As[16 * 136];
    __shared__ __align__(16) unsigned short Bs[128 * 136];

    const int tid     = threadIdx.x;
    const int rowbase = blockIdx.x * 16;                    // N=50000 divisible by 16

    // 1) kick off async DMA of all 128x128 bf16 weights into LDS.
    //    2048 chunks of 16B; 8 per thread. Tracked by ASYNCcnt.
    for (int i = tid; i < 128 * 16; i += 256) {
        const int r = i >> 4, c = i & 15;
        unsigned ldsa = (unsigned)(unsigned long long)(const void*)&Bs[r * 136 + c * 8];
        unsigned goff = (unsigned)(r * 256 + c * 16);       // byte offset into Wt
        asm volatile("global_load_async_to_lds_b128 %0, %1, %2"
                     :: "v"(ldsa), "v"(goff), "s"(Wt) : "memory");
    }

    // 2) overlap: stage A tile (16 x 128 f32 -> bf16) into LDS.
    //    X rows are streamed exactly once -> non-temporal.
    for (int i = tid; i < 16 * DF; i += 256) {
        const int r = i >> 7, c = i & 127;
        As[r * 136 + c] = f2bf_bits(
            __builtin_nontemporal_load(&X[(size_t)(rowbase + r) * DF + c]));
    }

    asm volatile("s_wait_asynccnt 0" ::: "memory");
    __syncthreads();

    const int wv   = tid >> 5;        // wave id == column tile == head
    const int lane = tid & 31;
    const int nl   = lane & 15;       // A: M row / B: N column held by this lane
    const int half = lane >> 4;       // K-half striping per ISA A/B layout

    // 3) preload all fragments (clause-batched ds_load_b128), then issue the
    //    four WMMAs back-to-back on the accumulation chain.
    v16bf afrag[4], bfrag[4];
    #pragma unroll
    for (int kk = 0; kk < 4; ++kk) {
        const int k0 = kk * 32;
        const u16x8* pa0 = (const u16x8*)&As[nl * 136 + k0 + half * 8];
        const u16x8* pa1 = (const u16x8*)&As[nl * 136 + k0 + 16 + half * 8];
        u16x16 ta = __builtin_shufflevector(pa0[0], pa1[0],
                     0,1,2,3,4,5,6,7,8,9,10,11,12,13,14,15);
        afrag[kk] = __builtin_bit_cast(v16bf, ta);
        const u16x8* pb0 = (const u16x8*)&Bs[(wv * 16 + nl) * 136 + k0 + half * 8];
        const u16x8* pb1 = (const u16x8*)&Bs[(wv * 16 + nl) * 136 + k0 + 16 + half * 8];
        u16x16 tb = __builtin_shufflevector(pb0[0], pb1[0],
                     0,1,2,3,4,5,6,7,8,9,10,11,12,13,14,15);
        bfrag[kk] = __builtin_bit_cast(v16bf, tb);
    }
    v8f acc = {};
    #pragma unroll
    for (int kk = 0; kk < 4; ++kk) {
        acc = __builtin_amdgcn_wmma_f32_16x16x32_bf16(
                  false, afrag[kk], false, bfrag[kk], (short)0, acc, false, false);
    }

    // 4) C/D layout: VGPR i -> M = i + 8*half, N = nl. Store h, reduce el/er.
    const float alv = al[wv * DH + nl];
    const float arv = ar[wv * DH + nl];
    #pragma unroll
    for (int i = 0; i < 8; ++i) {
        const int row = rowbase + i + 8 * half;
        float v = acc[i];
        hout[(size_t)row * DF + wv * DH + nl] = v;
        float pl = red16(v * alv);
        float pr = red16(v * arv);
        if (nl == 0) {
            el[(size_t)row * HH + wv] = pl;
            er[(size_t)row * HH + wv] = pr;
        }
    }
}

// ---------------------------------------------------------------------------
// Pass 1 over edges: logits + segment max (atomic) into mb[dst][h]
__global__ __launch_bounds__(256) void edge_logits_max(
    const int* __restrict__ src, const int* __restrict__ dst,
    const float* __restrict__ el, const float* __restrict__ er,
    float* __restrict__ elog, float* __restrict__ mb)
{
    int e = blockIdx.x * 256 + threadIdx.x;
    if (e >= EE) return;
    const int s = src[e], d = dst[e];
    #pragma unroll
    for (int h = 0; h < HH; ++h) {
        float v = el[(size_t)s * HH + h] + er[(size_t)d * HH + h];
        v = v > 0.0f ? v : LRELU_ATTN * v;
        elog[(size_t)e * HH + h] = v;
        atomicMaxF(&mb[(size_t)d * HH + h], v);
    }
}

// Pass 2: ex = exp(logit - m[dst]); z[dst] += ex (store ex in place of logit)
__global__ __launch_bounds__(256) void edge_exp_sum(
    const int* __restrict__ dst, const float* __restrict__ mb,
    float* __restrict__ elog, float* __restrict__ zb)
{
    int e = blockIdx.x * 256 + threadIdx.x;
    if (e >= EE) return;
    const int d = dst[e];
    #pragma unroll
    for (int h = 0; h < HH; ++h) {
        float ex = __expf(elog[(size_t)e * HH + h] - mb[(size_t)d * HH + h]);
        elog[(size_t)e * HH + h] = ex;
        atomicAdd(&zb[(size_t)d * HH + h], ex);
    }
}

// Pass 3: out[dst] += (ex/z[dst]) * h[src]   (one thread per (edge, head))
__global__ __launch_bounds__(256) void edge_aggregate(
    const int* __restrict__ src, const int* __restrict__ dst,
    const float* __restrict__ elog, const float* __restrict__ zb,
    const float* __restrict__ hbuf, float* __restrict__ out)
{
    int t = blockIdx.x * 256 + threadIdx.x;
    if (t >= EE * HH) return;
    const int e = t >> 3, h = t & 7;
    const int s = src[e], d = dst[e];
    const float a = elog[(size_t)e * HH + h] / zb[(size_t)d * HH + h];
    const float* hs = hbuf + (size_t)s * DF + h * DH;
    float*       po = out  + (size_t)d * DF + h * DH;
    #pragma unroll
    for (int i = 0; i < DH; ++i) atomicAdd(&po[i], a * hs[i]);
}

// Inter-layer activation (leaky_relu 0.01), in place
__global__ __launch_bounds__(256) void leaky_act(float* __restrict__ p, size_t n) {
    size_t i = (size_t)blockIdx.x * 256u + threadIdx.x;
    if (i < n) { float v = p[i]; p[i] = v > 0.0f ? v : LRELU_ACT * v; }
}

// ---------------------------------------------------------------------------
extern "C" void kernel_launch(void* const* d_in, const int* in_sizes, int n_in,
                              void* d_out, int out_size, void* d_ws, size_t ws_size,
                              hipStream_t stream) {
    const float* x   = (const float*)d_in[0];
    const int*   src = (const int*)  d_in[1];
    const int*   dst = (const int*)  d_in[2];
    const float* W1  = (const float*)d_in[3];
    const float* al1 = (const float*)d_in[4];
    const float* ar1 = (const float*)d_in[5];
    const float* W2  = (const float*)d_in[6];
    const float* al2 = (const float*)d_in[7];
    const float* ar2 = (const float*)d_in[8];
    float* out = (float*)d_out;

    // Workspace carve-up (all offsets 256B-aligned by construction)
    float* ws = (float*)d_ws;
    const size_t fh = (size_t)NN * DF;           // 6.4M floats
    const size_t nh = (size_t)NN * HH;           // 0.4M floats
    float* h1   = ws;                            // pre-attn features L1
    float* h2   = h1 + fh;                       // pre-attn features L2
    float* acc1 = h2 + fh;                       // L1 aggregated output / L2 input
    float* el   = acc1 + fh;
    float* er   = el + nh;
    float* mb   = er + nh;
    float* zb   = mb + nh;
    float* elog = zb + nh;                       // E*H logits -> exp
    unsigned short* wt1 = (unsigned short*)(elog + (size_t)EE * HH);
    unsigned short* wt2 = wt1 + DF * DF;

    const int EB  = (EE + 255) / 256;
    const int EHB = (EE * HH + 255) / 256;
    const int FB  = (int)((fh + 255) / 256);
    const int NHB = (int)((nh + 255) / 256);

    // Weight transpose + bf16 conversion (tiny)
    conv_wt<<<(DF * DF + 255) / 256, 256, 0, stream>>>(W1, wt1);
    conv_wt<<<(DF * DF + 255) / 256, 256, 0, stream>>>(W2, wt2);

    // ---------------- Layer 1 ----------------
    fill_u32<<<NHB, 256, 0, stream>>>((unsigned*)mb, nh, 0xFF800000u);  // -inf
    fill_u32<<<NHB, 256, 0, stream>>>((unsigned*)zb, nh, 0u);
    fill_u32<<<FB,  256, 0, stream>>>((unsigned*)acc1, fh, 0u);
    gat_gemm_wmma<<<NN / 16, 256, 0, stream>>>(x, wt1, al1, ar1, h1, el, er);
    edge_logits_max<<<EB, 256, 0, stream>>>(src, dst, el, er, elog, mb);
    edge_exp_sum<<<EB, 256, 0, stream>>>(dst, mb, elog, zb);
    edge_aggregate<<<EHB, 256, 0, stream>>>(src, dst, elog, zb, h1, acc1);
    leaky_act<<<FB, 256, 0, stream>>>(acc1, fh);

    // ---------------- Layer 2 ----------------
    fill_u32<<<NHB, 256, 0, stream>>>((unsigned*)mb, nh, 0xFF800000u);
    fill_u32<<<NHB, 256, 0, stream>>>((unsigned*)zb, nh, 0u);
    fill_u32<<<FB,  256, 0, stream>>>((unsigned*)out, fh, 0u);
    gat_gemm_wmma<<<NN / 16, 256, 0, stream>>>(acc1, wt2, al2, ar2, h2, el, er);
    edge_logits_max<<<EB, 256, 0, stream>>>(src, dst, el, er, elog, mb);
    edge_exp_sum<<<EB, 256, 0, stream>>>(dst, mb, elog, zb);
    edge_aggregate<<<EHB, 256, 0, stream>>>(src, dst, elog, zb, h2, out);
}
